// JointConvLayer_37134287242037
// MI455X (gfx1250) — matrix operand order repaired and checked by reading, not compiled, not run on hardware
//
#include <hip/hip_runtime.h>
#include <hip/hip_bf16.h>

// ---------------- problem constants (from reference) ----------------
#define N_ATOM_C 32768
#define N_RES_C  4096
#define F_ATOM_C 64
#define F_RES_C  128
#define S_C      9
#define H_EDGE_C 64
#define RES_H_C  128
#define E_BOND_C 65536
#define E_RAD_C  393216
#define E_ATOM_C (E_BOND_C + E_RAD_C)
#define E_RES_C  131072

// ---------------- WMMA types / helpers (CDNA5 gfx1250, wave32) ------
typedef __attribute__((ext_vector_type(16))) __bf16 v16bf;
typedef __attribute__((ext_vector_type(8)))  __bf16 v8bf;
typedef __attribute__((ext_vector_type(4)))  __bf16 v4bf;
typedef __attribute__((ext_vector_type(8)))  float  v8f;

static __device__ __forceinline__ __bf16 f2bf(float f) {
    union { float f; unsigned u; } in; in.f = f;
    unsigned r = (in.u + 0x7FFFu + ((in.u >> 16) & 1u)) >> 16;  // RNE
    union { unsigned short s; __bf16 b; } out; out.s = (unsigned short)r;
    return out.b;
}

static __device__ __forceinline__ v8f vzero8() {
    v8f z;
#pragma unroll
    for (int i = 0; i < 8; ++i) z[i] = 0.0f;
    return z;
}

// A-fragment (16x32 bf16) from row-major bf16 LDS tile [16][ld].
// ISA layout: lane half 0 -> K kb+0..7 / kb+16..23; half 1 -> K kb+8..15 / kb+24..31.
// Two contiguous 16-byte runs -> 2x ds_load_b128.
static __device__ __forceinline__ v16bf load_afrag(const __bf16* m, int ld, int kb,
                                                   int lrow, int lhalf) {
    const __bf16* row = m + lrow * ld + kb + lhalf * 8;
    v8bf lo = *(const v8bf*)(row);
    v8bf hi = *(const v8bf*)(row + 16);
    return __builtin_shufflevector(lo, hi, 0, 1, 2, 3, 4, 5, 6, 7,
                                   8, 9, 10, 11, 12, 13, 14, 15);
}

// B-fragment from pre-swizzled packed weights: fragment (kt,nt) occupies 512
// bf16 (32 lanes x 16); each lane loads its contiguous 32 bytes.
static __device__ __forceinline__ v16bf load_bfrag(const __bf16* __restrict__ packed,
                                                   int NT, int kt, int nt, int lane) {
    return *(const v16bf*)(packed + (((size_t)(kt * NT + nt)) << 9) + (lane << 4));
}

static __device__ __forceinline__ v8f wmma_bf(v16bf a, v16bf b, v8f c) {
    return __builtin_amdgcn_wmma_f32_16x16x32_bf16(false, a, false, b,
                                                   (short)0, c, false, false);
}

// ---------------- weight pre-pack: f32 [K][N] -> bf16 B-fragments ----
// dst[((kt*NT + nt)*32 + lane)*16 + i] = bf16(src[(kt*32 + (lane>>4)*8 + (i&7) + (i>>3)*16)][nt*16 + (lane&15)])
__global__ void pack_bfrag_kernel(const float* __restrict__ src,
                                  __bf16* __restrict__ dst, int K, int N) {
    int idx = blockIdx.x * blockDim.x + threadIdx.x;
    if (idx >= K * N) return;
    int i    = idx & 15;
    int lane = (idx >> 4) & 31;
    int ft   = idx >> 9;
    int NT   = N >> 4;
    int nt   = ft % NT;
    int kt   = ft / NT;
    int k = kt * 32 + (lane >> 4) * 8 + (i & 7) + (i >> 3) * 16;
    int n = nt * 16 + (lane & 15);
    dst[idx] = f2bf(src[(size_t)k * N + n]);
}

// ---------------- fused edge-conv kernel ----------------------------
// Per 16-edge tile (one wave, intra-wave LDS exchange only):
//   h   = relu(efeat @ W1 + b1)                    [16 x H]
//   for each 32-wide chunk kc of WN = F_IN*9:
//     wc  = h @ W2[:,kc] + b2[kc]                  [16 x 32]   (C-layout regs)
//     pc  = wc * (x_src outer sh)[.,kc]            elementwise, bf16 LDS bounce
//     msg += pc @ Wout[kc,:]                       [16 x F_OUT] (K-step WMMA)
//   atomicAdd msg into acc[dst], atomicAdd 1 into cnt[dst]
template <int H, int F_IN, int F_OUT>
__launch_bounds__(64)
__global__ void fused_conv_kernel(const float* __restrict__ efeat,
                                  const __bf16* __restrict__ W1p, const float* __restrict__ b1,
                                  const __bf16* __restrict__ W2p, const float* __restrict__ b2,
                                  const __bf16* __restrict__ Woutp,
                                  const float* __restrict__ xsrc,
                                  const int*  __restrict__ src_idx,   // nullptr -> identity
                                  const float* __restrict__ sh,
                                  const int*  __restrict__ dst_idx,
                                  float* __restrict__ acc,
                                  float* __restrict__ cnt,
                                  int e_base) {
    constexpr int WN = F_IN * S_C;
    // per-wave LDS byte layout (all sub-arrays 32B-aligned)
    constexpr int FEAT_B = 16 * H * 2;       // bf16 [16][H]
    constexpr int P_B    = 16 * 32 * 2;      // bf16 [16][32]
    constexpr int X_B    = 16 * F_IN * 4;    // f32  [16][F_IN]
    constexpr int SH_B   = 16 * S_C * 4;     // f32  [16][9]
    constexpr int PW_B   = FEAT_B * 2 + P_B + X_B + SH_B;
    __shared__ __align__(32) char smem_raw[2 * PW_B + H * 4 + WN * 4];

    const int wave  = threadIdx.x >> 5;
    const int lane  = threadIdx.x & 31;
    const int lhalf = lane >> 4;
    const int lrow  = lane & 15;

    char*   base   = smem_raw + wave * PW_B;
    __bf16* s_feat = (__bf16*)base;                    // [16][H]
    __bf16* s_h    = s_feat + 16 * H;                  // [16][H]
    __bf16* s_p    = s_h + 16 * H;                     // [16][32]
    float*  s_x    = (float*)(base + FEAT_B * 2 + P_B);// [16][F_IN]
    float*  s_sh   = s_x + 16 * F_IN;                  // [16][9]
    float*  s_b1   = (float*)(smem_raw + 2 * PW_B);    // [H]
    float*  s_b2   = s_b1 + H;                         // [WN]

    const int tile = blockIdx.x * 2 + wave;
    const int e0   = tile * 16;          // group-local edge base (features)
    const int g0   = e_base + e0;        // global edge base (sh / src / dst)

    // ---- stage inputs into LDS (vectorized: b128 loads, 8B bf16 stores) ----
    // 16 consecutive feature rows are contiguous in memory.
    for (int p = lane * 4; p < 16 * H; p += 128) {
        const float4 f = *(const float4*)(efeat + (size_t)e0 * H + p);
        v4bf o;
        o[0] = f2bf(f.x); o[1] = f2bf(f.y); o[2] = f2bf(f.z); o[3] = f2bf(f.w);
        *(v4bf*)(s_feat + p) = o;
    }
    // gathered x_src rows: 4 consecutive cols per lane-chunk
    for (int c = lane; c < 16 * (F_IN / 4); c += 32) {
        int m   = c / (F_IN / 4);
        int off = (c % (F_IN / 4)) * 4;
        int srow = src_idx ? src_idx[g0 + m] : (g0 + m);
        *(float4*)(s_x + m * F_IN + off) =
            *(const float4*)(xsrc + (size_t)srow * F_IN + off);
    }
    for (int idx = lane; idx < 16 * S_C; idx += 32)
        s_sh[idx] = sh[(size_t)(g0 + idx / S_C) * S_C + (idx % S_C)];
    // biases (whole block cooperates -> needs the one block barrier below)
    for (int idx = threadIdx.x; idx < H; idx += 64)  s_b1[idx] = b1[idx];
    for (int idx = threadIdx.x; idx < WN; idx += 64) s_b2[idx] = b2[idx];
    __syncthreads();

    // ---- MLP layer 1: h = relu(feat @ W1 + b1) ----
#pragma unroll
    for (int nt = 0; nt < H / 16; ++nt) {
        v8f c = vzero8();
#pragma unroll
        for (int kc = 0; kc < H / 32; ++kc)
            c = wmma_bf(load_afrag(s_feat, H, kc * 32, lrow, lhalf),
                        load_bfrag(W1p, H / 16, kc, nt, lane), c);
#pragma unroll
        for (int r = 0; r < 8; ++r) {
            int m = r + lhalf * 8;
            int n = nt * 16 + lrow;
            float v = c[r] + s_b1[n];
            s_h[m * H + n] = f2bf(v > 0.0f ? v : 0.0f);
        }
    }
    // intra-wave LDS exchange only: LDS is in-order per wave, just fence the
    // compiler's scheduler (no hardware barrier needed).
    __builtin_amdgcn_wave_barrier();

    // ---- fused MLP2 + outer-product + projection over K chunks ----
    v8f macc[F_OUT / 16];
#pragma unroll
    for (int nt = 0; nt < F_OUT / 16; ++nt) macc[nt] = vzero8();

    for (int kc = 0; kc < WN / 32; ++kc) {
#pragma unroll
        for (int t = 0; t < 2; ++t) {
            v8f c = vzero8();
#pragma unroll
            for (int kh = 0; kh < H / 32; ++kh)
                c = wmma_bf(load_afrag(s_h, H, kh * 32, lrow, lhalf),
                            load_bfrag(W2p, WN / 16, kh, kc * 2 + t, lane), c);
            int nl = t * 16 + lrow;
            int fi = kc * 32 + nl;
            int f = fi / S_C, si = fi - f * S_C;
            float bv = s_b2[fi];
#pragma unroll
            for (int r = 0; r < 8; ++r) {
                int m  = r + lhalf * 8;
                float o = s_x[m * F_IN + f] * s_sh[m * S_C + si];
                s_p[m * 32 + nl] = f2bf((c[r] + bv) * o);
            }
        }
        __builtin_amdgcn_wave_barrier();   // stores above -> loads below (in-order LDS)
        v16bf a = load_afrag(s_p, 32, 0, lrow, lhalf);
#pragma unroll
        for (int nt = 0; nt < F_OUT / 16; ++nt)
            macc[nt] = wmma_bf(a, load_bfrag(Woutp, F_OUT / 16, kc, nt, lane),
                               macc[nt]);
        __builtin_amdgcn_wave_barrier();   // loads above -> next iter's stores (WAR)
    }

    // ---- scatter-add messages + counts ----
#pragma unroll
    for (int r = 0; r < 8; ++r) {
        int m = r + lhalf * 8;
        int d = dst_idx[g0 + m];
#pragma unroll
        for (int nt = 0; nt < F_OUT / 16; ++nt)
            atomicAdd(&acc[(size_t)d * F_OUT + nt * 16 + lrow], macc[nt][r]);
    }
    if (lane < 16) atomicAdd(&cnt[dst_idx[g0 + lane]], 1.0f);
}

// ---------------- elementwise helpers -------------------------------
__global__ void zero_kernel(float* __restrict__ p, int n) {
    int i = blockIdx.x * blockDim.x + threadIdx.x;
    if (i < n) p[i] = 0.0f;
}

// out = acc / max(cnt,1) + residual
__global__ void finalize_kernel(const float* __restrict__ acc,
                                const float* __restrict__ cnt,
                                const float* __restrict__ residual,
                                float* __restrict__ out, int rows, int cols) {
    int i = blockIdx.x * blockDim.x + threadIdx.x;
    if (i >= rows * cols) return;
    float c = cnt[i / cols];
    c = c > 1.0f ? c : 1.0f;
    out[i] = acc[i] / c + residual[i];
}

// ---------------- launch ---------------------------------------------
extern "C" void kernel_launch(void* const* d_in, const int* in_sizes, int n_in,
                              void* d_out, int out_size, void* d_ws, size_t ws_size,
                              hipStream_t stream) {
    const float* atom_features        = (const float*)d_in[0];
    const float* res_features         = (const float*)d_in[1];
    const int*   atom_edge_index      = (const int*)  d_in[2];   // [2, E_ATOM]
    const float* bond_features        = (const float*)d_in[3];
    const float* radius_edge_features = (const float*)d_in[4];
    const float* atom_edge_sh         = (const float*)d_in[5];
    const int*   atom_res_batch       = (const int*)  d_in[6];
    const float* agg_edge_features    = (const float*)d_in[7];
    const float* agg_edge_sh          = (const float*)d_in[8];
    const int*   res_edge_index       = (const int*)  d_in[9];   // [2, E_RES]
    const float* res_edge_features    = (const float*)d_in[10];
    const float* res_edge_sh          = (const float*)d_in[11];
    const float* Wb1 = (const float*)d_in[12]; const float* bb1 = (const float*)d_in[13];
    const float* Wb2 = (const float*)d_in[14]; const float* bb2 = (const float*)d_in[15];
    const float* Wr1 = (const float*)d_in[16]; const float* br1 = (const float*)d_in[17];
    const float* Wr2 = (const float*)d_in[18]; const float* br2 = (const float*)d_in[19];
    const float* W_atom_out = (const float*)d_in[20];
    const float* Wa1 = (const float*)d_in[21]; const float* ba1 = (const float*)d_in[22];
    const float* Wa2 = (const float*)d_in[23]; const float* ba2 = (const float*)d_in[24];
    const float* W_agg_out = (const float*)d_in[25];
    const float* Ws1 = (const float*)d_in[26]; const float* bs1 = (const float*)d_in[27];
    const float* Ws2 = (const float*)d_in[28]; const float* bs2 = (const float*)d_in[29];
    const float* W_res_out = (const float*)d_in[30];

    const int* a_src = atom_edge_index;
    const int* a_dst = atom_edge_index + E_ATOM_C;
    const int* r_src = res_edge_index;
    const int* r_dst = res_edge_index + E_RES_C;

    // ---- workspace layout: f32 accumulators, then bf16 packed weights ----
    float* ws = (float*)d_ws;
    size_t A_ACC  = 0;
    size_t A_CNT  = A_ACC  + (size_t)N_ATOM_C * F_ATOM_C;
    size_t RA_ACC = A_CNT  + N_ATOM_C;
    size_t RA_CNT = RA_ACC + (size_t)N_RES_C * F_RES_C;
    size_t R_MID  = RA_CNT + N_RES_C;
    size_t RB_ACC = R_MID  + (size_t)N_RES_C * F_RES_C;
    size_t RB_CNT = RB_ACC + (size_t)N_RES_C * F_RES_C;
    size_t WS_F32 = (RB_CNT + N_RES_C + 31) & ~(size_t)31;

    __bf16* pw = (__bf16*)(ws + WS_F32);
    size_t pofs = 0;
    auto alloc_bf = [&](size_t n) {
        __bf16* p = pw + pofs;
        pofs = (pofs + n + 31) & ~(size_t)31;
        return p;
    };
    const int WN_A = F_ATOM_C * S_C;   // 576
    const int WN_R = F_RES_C * S_C;    // 1152
    __bf16* Wb1p = alloc_bf((size_t)H_EDGE_C * H_EDGE_C);
    __bf16* Wb2p = alloc_bf((size_t)H_EDGE_C * WN_A);
    __bf16* Wr1p = alloc_bf((size_t)H_EDGE_C * H_EDGE_C);
    __bf16* Wr2p = alloc_bf((size_t)H_EDGE_C * WN_A);
    __bf16* Wa1p = alloc_bf((size_t)H_EDGE_C * H_EDGE_C);
    __bf16* Wa2p = alloc_bf((size_t)H_EDGE_C * WN_A);
    __bf16* Waop = alloc_bf((size_t)WN_A * F_ATOM_C);   // W_atom_out
    __bf16* Wgop = alloc_bf((size_t)WN_A * F_RES_C);    // W_agg_out
    __bf16* Ws1p = alloc_bf((size_t)RES_H_C * RES_H_C);
    __bf16* Ws2p = alloc_bf((size_t)RES_H_C * WN_R);
    __bf16* Wrop = alloc_bf((size_t)WN_R * F_RES_C);    // W_res_out

    float* atom_out = (float*)d_out;                                   // [N_ATOM, 64]
    float* res_out  = (float*)d_out + (size_t)N_ATOM_C * F_ATOM_C;     // [N_RES, 128]

    // 0a) zero accumulators / counts
    {
        int n = (int)WS_F32;
        zero_kernel<<<(n + 255) / 256, 256, 0, stream>>>(ws, n);
    }
    // 0b) pre-pack weights into bf16 B-fragment layout (one-time per launch)
    auto pack = [&](const float* src, __bf16* dst, int K, int N) {
        int n = K * N;
        pack_bfrag_kernel<<<(n + 255) / 256, 256, 0, stream>>>(src, dst, K, N);
    };
    pack(Wb1, Wb1p, H_EDGE_C, H_EDGE_C);
    pack(Wb2, Wb2p, H_EDGE_C, WN_A);
    pack(Wr1, Wr1p, H_EDGE_C, H_EDGE_C);
    pack(Wr2, Wr2p, H_EDGE_C, WN_A);
    pack(Wa1, Wa1p, H_EDGE_C, H_EDGE_C);
    pack(Wa2, Wa2p, H_EDGE_C, WN_A);
    pack(W_atom_out, Waop, WN_A, F_ATOM_C);
    pack(W_agg_out,  Wgop, WN_A, F_RES_C);
    pack(Ws1, Ws1p, RES_H_C, RES_H_C);
    pack(Ws2, Ws2p, RES_H_C, WN_R);
    pack(W_res_out, Wrop, WN_R, F_RES_C);

    // 1) atom graph conv: bond edges then radius edges (shared accumulators)
    fused_conv_kernel<H_EDGE_C, F_ATOM_C, F_ATOM_C>
        <<<E_BOND_C / 32, 64, 0, stream>>>(
            bond_features, Wb1p, bb1, Wb2p, bb2, Waop,
            atom_features, a_src, atom_edge_sh, a_dst,
            ws + A_ACC, ws + A_CNT, 0);
    fused_conv_kernel<H_EDGE_C, F_ATOM_C, F_ATOM_C>
        <<<E_RAD_C / 32, 64, 0, stream>>>(
            radius_edge_features, Wr1p, br1, Wr2p, br2, Waop,
            atom_features, a_src, atom_edge_sh, a_dst,
            ws + A_ACC, ws + A_CNT, E_BOND_C);

    // 2) atom_out = scatter_mean + residual
    {
        int n = N_ATOM_C * F_ATOM_C;
        finalize_kernel<<<(n + 255) / 256, 256, 0, stream>>>(
            ws + A_ACC, ws + A_CNT, atom_features, atom_out, N_ATOM_C, F_ATOM_C);
    }

    // 3) atom -> residue aggregation conv (identity src: one edge per atom)
    fused_conv_kernel<H_EDGE_C, F_ATOM_C, F_RES_C>
        <<<N_ATOM_C / 32, 64, 0, stream>>>(
            agg_edge_features, Wa1p, ba1, Wa2p, ba2, Wgop,
            atom_out, nullptr, agg_edge_sh, atom_res_batch,
            ws + RA_ACC, ws + RA_CNT, 0);

    // 4) res_mid = scatter_mean + residual
    {
        int n = N_RES_C * F_RES_C;
        finalize_kernel<<<(n + 255) / 256, 256, 0, stream>>>(
            ws + RA_ACC, ws + RA_CNT, res_features, ws + R_MID, N_RES_C, F_RES_C);
    }

    // 5) residue graph conv
    fused_conv_kernel<RES_H_C, F_RES_C, F_RES_C>
        <<<E_RES_C / 32, 64, 0, stream>>>(
            res_edge_features, Ws1p, bs1, Ws2p, bs2, Wrop,
            ws + R_MID, r_src, res_edge_sh, r_dst,
            ws + RB_ACC, ws + RB_CNT, 0);

    // 6) res_out = scatter_mean + residual
    {
        int n = N_RES_C * F_RES_C;
        finalize_kernel<<<(n + 255) / 256, 256, 0, stream>>>(
            ws + RB_ACC, ws + RB_CNT, ws + R_MID, res_out, N_RES_C, F_RES_C);
    }
}
